// IMFF_46351287059052
// MI455X (gfx1250) — matrix operand order repaired and checked by reading, not compiled, not run on hardware
//
#include <hip/hip_runtime.h>
#include <math.h>

// ---------------------------------------------------------------------------
// Types for CDNA5 WMMA
// ---------------------------------------------------------------------------
typedef __attribute__((ext_vector_type(16))) __bf16 v16bf;
typedef __attribute__((ext_vector_type(4)))  __bf16 v4bf;
typedef __attribute__((ext_vector_type(8)))  float  v8f;

#define TM 128
#define TN 128
#define TK 32
#define LDS_STRIDE 40   // bf16 elements; 32 data + 8 pad (80B rows, 8B aligned)

// ---------------------------------------------------------------------------
// Generic GEMM:  Y[b] (M x N) = alpha * X[b] (M x K) @ op(W[b]) + bias
//   TRANSB == false : W is (N x K) row-major ("@ W.T" convention)
//   TRANSB == true  : W is (K x N) row-major (plain X @ W)
// Y written at  Yb + m*ymStride + n*ynStride  (allows transposed output).
// f32 in global, bf16 staged in LDS (float4 loads -> packed b64 stores),
// f32 WMMA accumulate. Dims: M%128==0, N%128==0, K%32==0 (true for all uses).
// Block = 256 threads = 8 waves = 4 M-waves x 2 N-waves; each wave:
//   2 A-frags x 4 B-frags -> 8 x v_wmma_f32_16x16x32_bf16 per K-step.
// ---------------------------------------------------------------------------
template <bool TRANSB>
__global__ __launch_bounds__(256) void imff_gemm_wmma(
    const float* __restrict__ X, long long xBatch,
    const float* __restrict__ W, long long wBatch,
    const float* __restrict__ bias,
    float* __restrict__ Y, long long yBatch,
    int M, int N, int K,
    int ymStride, int ynStride, float alpha)
{
    __shared__ __bf16 Xs[TM * LDS_STRIDE];
    __shared__ __bf16 Ws[TN * LDS_STRIDE];

    const int b  = blockIdx.z;
    const float* Xb = X + (long long)b * xBatch;
    const float* Wb = W + (long long)b * wBatch;
    float*       Yb = Y + (long long)b * yBatch;

    const int m0 = blockIdx.y * TM;
    const int n0 = blockIdx.x * TN;

    const int t     = threadIdx.x;
    const int lane  = t & 31;
    const int wv    = t >> 5;          // 0..7
    const int mwave = wv & 3;          // 4 waves along M (32 rows each)
    const int nwave = wv >> 2;         // 2 waves along N (64 cols each)
    const int half  = (lane >> 4) & 1;
    const int lr    = lane & 15;

    v8f acc[2][4];
    const v8f vzero = {0.f, 0.f, 0.f, 0.f, 0.f, 0.f, 0.f, 0.f};
#pragma unroll
    for (int a = 0; a < 2; ++a)
#pragma unroll
        for (int j = 0; j < 4; ++j) acc[a][j] = vzero;

    for (int k0 = 0; k0 < K; k0 += TK) {
        // ---- stage X tile (TM x TK): 1024 float4 chunks, 4 per thread ----
        float4 xv[4];
#pragma unroll
        for (int i = 0; i < 4; ++i) {
            int chunk = i * 256 + t;
            int r = chunk >> 3, c4 = (chunk & 7) << 2;
            xv[i] = *(const float4*)&Xb[(long long)(m0 + r) * K + k0 + c4];
        }
        // ---- stage W tile (TN x TK logical, stored as Ws[n][k]) ----------
        float4 wv4[4];
#pragma unroll
        for (int i = 0; i < 4; ++i) {
            int chunk = i * 256 + t;
            if (TRANSB) {
                int kk = chunk >> 5, nc = (chunk & 31) << 2;
                wv4[i] = *(const float4*)&Wb[(long long)(k0 + kk) * N + n0 + nc];
            } else {
                int r = chunk >> 3, c4 = (chunk & 7) << 2;
                wv4[i] = *(const float4*)&Wb[(long long)(n0 + r) * K + k0 + c4];
            }
        }
        if (k0 + TK < K) {  // prefetch next X tile line for this thread
            __builtin_prefetch(&Xb[(long long)(m0 + (t >> 3)) * K + k0 + TK + ((t & 7) << 2)], 0, 3);
        }
#pragma unroll
        for (int i = 0; i < 4; ++i) {
            int chunk = i * 256 + t;
            int r = chunk >> 3, c4 = (chunk & 7) << 2;
            v4bf p;
            p[0] = (__bf16)xv[i].x; p[1] = (__bf16)xv[i].y;
            p[2] = (__bf16)xv[i].z; p[3] = (__bf16)xv[i].w;
            *(v4bf*)&Xs[r * LDS_STRIDE + c4] = p;
        }
#pragma unroll
        for (int i = 0; i < 4; ++i) {
            int chunk = i * 256 + t;
            if (TRANSB) {
                int kk = chunk >> 5, nc = (chunk & 31) << 2;
                Ws[(nc + 0) * LDS_STRIDE + kk] = (__bf16)wv4[i].x;
                Ws[(nc + 1) * LDS_STRIDE + kk] = (__bf16)wv4[i].y;
                Ws[(nc + 2) * LDS_STRIDE + kk] = (__bf16)wv4[i].z;
                Ws[(nc + 3) * LDS_STRIDE + kk] = (__bf16)wv4[i].w;
            } else {
                int r = chunk >> 3, c4 = (chunk & 7) << 2;
                v4bf p;
                p[0] = (__bf16)wv4[i].x; p[1] = (__bf16)wv4[i].y;
                p[2] = (__bf16)wv4[i].z; p[3] = (__bf16)wv4[i].w;
                *(v4bf*)&Ws[r * LDS_STRIDE + c4] = p;
            }
        }
        __syncthreads();

        // ---- A fragments: 16x32 bf16, rows mwave*32 + a*16 ---------------
        v16bf afrag[2];
#pragma unroll
        for (int a = 0; a < 2; ++a) {
            const __bf16* row = &Xs[(mwave * 32 + a * 16 + lr) * LDS_STRIDE];
#pragma unroll
            for (int e = 0; e < 16; ++e) {
                int kk = ((e >> 3) << 4) + (half << 3) + (e & 7);
                afrag[a][e] = row[kk];
            }
        }
        // ---- B fragments: cols nwave*64 + j*16 ---------------------------
        v16bf bfrag[4];
#pragma unroll
        for (int j = 0; j < 4; ++j) {
            const __bf16* row = &Ws[(nwave * 64 + j * 16 + lr) * LDS_STRIDE];
#pragma unroll
            for (int e = 0; e < 16; ++e)
                bfrag[j][e] = row[((e >> 3) << 4) + (half << 3) + (e & 7)];
        }
        // ---- 8 WMMAs ------------------------------------------------------
#pragma unroll
        for (int a = 0; a < 2; ++a)
#pragma unroll
            for (int j = 0; j < 4; ++j)
                acc[a][j] = __builtin_amdgcn_wmma_f32_16x16x32_bf16(
                    false, afrag[a], false, bfrag[j], (short)0, acc[a][j], false, false);
        __syncthreads();
    }

    // ---- epilogue: D layout = VGPR v -> M = v + half*8, N = lr -----------
#pragma unroll
    for (int a = 0; a < 2; ++a) {
#pragma unroll
        for (int j = 0; j < 4; ++j) {
            int n = n0 + nwave * 64 + j * 16 + lr;
            float bv = bias ? bias[n] : 0.f;
#pragma unroll
            for (int v = 0; v < 8; ++v) {
                int m = m0 + mwave * 32 + a * 16 + v + half * 8;
                Yb[(long long)m * ymStride + (long long)n * ynStride] =
                    acc[a][j][v] * alpha + bv;
            }
        }
    }
}

// ---------------------------------------------------------------------------
// Stage 1: per batch, q1_0 = q1_w @ l1[:,0]; w_eff = k1_w^T @ q1_0
// ---------------------------------------------------------------------------
__global__ __launch_bounds__(256) void imff_stage1(
    const float* __restrict__ l1, const float* __restrict__ q1w,
    const float* __restrict__ k1w, float* __restrict__ weff,
    int Cin, int Cout, int N1)
{
    __shared__ float l1col[256];
    __shared__ float q10[512];
    const int b = blockIdx.x;
    const int t = threadIdx.x;

    l1col[t] = l1[((long long)b * Cin + t) * N1];   // column 0
    __syncthreads();

    for (int o = t; o < Cout; o += 256) {
        const float* wr = q1w + (long long)o * Cin;
        float s = 0.f;
        for (int c = 0; c < Cin; ++c) s += wr[c] * l1col[c];
        q10[o] = s;
    }
    __syncthreads();

    float s = 0.f;
    for (int o = 0; o < Cout; ++o) s += k1w[(long long)o * Cin + t] * q10[o];
    weff[b * Cin + t] = s;
}

// scores[b,n] = dot(weff[b], l1[b,:,n])
__global__ __launch_bounds__(256) void imff_stage2(
    const float* __restrict__ l1, const float* __restrict__ weff,
    float* __restrict__ scores, int Cin, int N1)
{
    long long g = (long long)blockIdx.x * 256 + threadIdx.x; // b*N1 + n
    int b = (int)(g / N1), n = (int)(g % N1);
    const float* lb = l1 + (long long)b * Cin * N1 + n;
    const float* w  = weff + b * Cin;
    float s = 0.f;
    for (int c = 0; c < Cin; ++c) s += w[c] * lb[(long long)c * N1];
    scores[g] = s;
}

// ---------------------------------------------------------------------------
// Top-k via full in-LDS bitonic sort of 4096 64-bit keys, descending.
// Key = (order-preserving f32 map << 32) | (N-1-idx)  => ties pick lower idx.
// ---------------------------------------------------------------------------
__global__ __launch_bounds__(1024) void imff_topk(
    const float* __restrict__ scores, int* __restrict__ idxo, int N, int Kout)
{
    __shared__ unsigned long long key[4096];
    const int b = blockIdx.x;
    const float* s = scores + (long long)b * N;

    for (int i = threadIdx.x; i < N; i += 1024) {
        unsigned u = __float_as_uint(s[i]);
        u = (u & 0x80000000u) ? ~u : (u | 0x80000000u);
        key[i] = ((unsigned long long)u << 32) | (unsigned)(N - 1 - i);
    }
    __syncthreads();

    for (int k = 2; k <= N; k <<= 1) {
        for (int j = k >> 1; j > 0; j >>= 1) {
            for (int i = threadIdx.x; i < N; i += 1024) {
                int ixj = i ^ j;
                if (ixj > i) {
                    bool desc = ((i & k) == 0);
                    unsigned long long a = key[i], c = key[ixj];
                    if (desc ? (a < c) : (a > c)) { key[i] = c; key[ixj] = a; }
                }
            }
            __syncthreads();
        }
    }
    if (threadIdx.x < Kout)
        idxo[b * Kout + threadIdx.x] =
            (N - 1) - (int)(unsigned)(key[threadIdx.x] & 0xFFFFFFFFu);
}

// Xsel[b][n][c] = l1[b][c][idx[b][n]]   (token-major gather)
__global__ void imff_gather_l1(const float* __restrict__ l1,
                               const int* __restrict__ idx,
                               float* __restrict__ Xsel,
                               int Cin, int N1, int Ksel)
{
    long long g = (long long)blockIdx.x * 256 + threadIdx.x;
    int c = (int)(g % Cin);
    long long t2 = g / Cin;
    int n = (int)(t2 % Ksel), b = (int)(t2 / Ksel);
    Xsel[g] = l1[((long long)b * Cin + c) * N1 + idx[b * Ksel + n]];
}

// p1sel[b][j][n] = xyz1[b][j][idx[b][n]]
__global__ void imff_gather_xyz(const float* __restrict__ xyz1,
                                const int* __restrict__ idx,
                                float* __restrict__ p1sel, int N1, int Ksel)
{
    long long g = (long long)blockIdx.x * 256 + threadIdx.x;
    int n = (int)(g % Ksel);
    long long t2 = g / Ksel;
    int j = (int)(t2 % 3), b = (int)(t2 / 3);
    p1sel[g] = xyz1[((long long)b * 3 + j) * N1 + idx[b * Ksel + n]];
}

// qin[b][n][o] = l2[b][o][n] + posw[o]·xyz2[b][:,n]
__global__ __launch_bounds__(256) void imff_build_qin(
    const float* __restrict__ l2, const float* __restrict__ xyz2,
    const float* __restrict__ posw, float* __restrict__ qin,
    int Cout, int N2)
{
    int bn = blockIdx.x, b = bn / N2, n = bn % N2;
    const float* xz = xyz2 + (long long)b * 3 * N2 + n;
    float px = xz[0], py = xz[N2], pz = xz[2 * N2];
    const float* l2b = l2 + (long long)b * Cout * N2 + n;
    float* out = qin + (long long)bn * Cout;
    for (int o = threadIdx.x; o < Cout; o += 256) {
        const float* pw = posw + o * 3;
        out[o] = l2b[(long long)o * N2] + pw[0] * px + pw[1] * py + pw[2] * pz;
    }
}

// kin[b][n][o] = x2[b][n][o] + posw[o]·p1sel[b][:,n]
__global__ __launch_bounds__(256) void imff_build_kin(
    const float* __restrict__ x2, const float* __restrict__ p1sel,
    const float* __restrict__ posw, float* __restrict__ kin,
    int Cout, int N2)
{
    int bn = blockIdx.x, b = bn / N2, n = bn % N2;
    const float* pv = p1sel + (long long)b * 3 * N2 + n;
    float px = pv[0], py = pv[N2], pz = pv[2 * N2];
    const float* xr = x2 + (long long)bn * Cout;
    float* out = kin + (long long)bn * Cout;
    for (int o = threadIdx.x; o < Cout; o += 256) {
        const float* pw = posw + o * 3;
        out[o] = xr[o] + pw[0] * px + pw[1] * py + pw[2] * pz;
    }
}

// ---------------------------------------------------------------------------
// LayerNorm over last dim C (one block per token row), optional residual read
// from (B,C,Mtok) layout, optional activation (0 none, 1 relu, 2 leaky 0.01)
// ---------------------------------------------------------------------------
__global__ __launch_bounds__(256) void imff_ln_act(
    float* __restrict__ x, const float* __restrict__ resid,
    const float* __restrict__ g, const float* __restrict__ be,
    int C, int Mtok, int act)
{
    __shared__ float red[256];
    const int row = blockIdx.x;
    const int b = row / Mtok, n = row % Mtok;
    float* xr = x + (long long)row * C;
    const float* rr = resid ? resid + (long long)b * C * Mtok + n : nullptr;

    float s = 0.f;
    for (int c = threadIdx.x; c < C; c += 256) {
        float v = xr[c];
        if (rr) { v += rr[(long long)c * Mtok]; xr[c] = v; }
        s += v;
    }
    red[threadIdx.x] = s; __syncthreads();
    for (int o = 128; o > 0; o >>= 1) {
        if (threadIdx.x < o) red[threadIdx.x] += red[threadIdx.x + o];
        __syncthreads();
    }
    float mean = red[0] / C; __syncthreads();

    float s2 = 0.f;
    for (int c = threadIdx.x; c < C; c += 256) {
        float d = xr[c] - mean; s2 += d * d;
    }
    red[threadIdx.x] = s2; __syncthreads();
    for (int o = 128; o > 0; o >>= 1) {
        if (threadIdx.x < o) red[threadIdx.x] += red[threadIdx.x + o];
        __syncthreads();
    }
    float inv = rsqrtf(red[0] / C + 1e-5f);

    for (int c = threadIdx.x; c < C; c += 256) {
        float v = (xr[c] - mean) * inv * g[c] + be[c];
        if (act == 1)      v = fmaxf(v, 0.f);
        else if (act == 2) v = (v > 0.f) ? v : 0.01f * v;
        xr[c] = v;
    }
}

// Row softmax, in place, one block per row of length L
__global__ __launch_bounds__(256) void imff_softmax(float* __restrict__ S, int L)
{
    __shared__ float red[256];
    float* r = S + (long long)blockIdx.x * L;
    float mx = -3.0e38f;
    for (int i = threadIdx.x; i < L; i += 256) mx = fmaxf(mx, r[i]);
    red[threadIdx.x] = mx; __syncthreads();
    for (int o = 128; o > 0; o >>= 1) {
        if (threadIdx.x < o) red[threadIdx.x] = fmaxf(red[threadIdx.x], red[threadIdx.x + o]);
        __syncthreads();
    }
    mx = red[0]; __syncthreads();
    float s = 0.f;
    for (int i = threadIdx.x; i < L; i += 256) {
        float e = __expf(r[i] - mx); r[i] = e; s += e;
    }
    red[threadIdx.x] = s; __syncthreads();
    for (int o = 128; o > 0; o >>= 1) {
        if (threadIdx.x < o) red[threadIdx.x] += red[threadIdx.x + o];
        __syncthreads();
    }
    float inv = 1.0f / red[0];
    for (int i = threadIdx.x; i < L; i += 256) r[i] *= inv;
}

// ---------------------------------------------------------------------------
extern "C" void kernel_launch(void* const* d_in, const int* in_sizes, int n_in,
                              void* d_out, int out_size, void* d_ws, size_t ws_size,
                              hipStream_t stream)
{
    (void)in_sizes; (void)n_in; (void)out_size; (void)ws_size;
    const int B = 8, Cin = 256, Cout = 512, Rout = 256, N1 = 4096, N2 = 1024;

    const float* l1     = (const float*)d_in[0];
    const float* xyz1   = (const float*)d_in[1];
    const float* l2     = (const float*)d_in[2];
    const float* xyz2   = (const float*)d_in[3];
    const float* q1_w   = (const float*)d_in[4];
    const float* k1_w   = (const float*)d_in[5];
    const float* mlp_w1 = (const float*)d_in[6];
    const float* mlp_b1 = (const float*)d_in[7];
    const float* mlp_g1 = (const float*)d_in[8];
    const float* mlp_be1= (const float*)d_in[9];
    const float* mlp_w2 = (const float*)d_in[10];
    const float* mlp_b2 = (const float*)d_in[11];
    const float* qw     = (const float*)d_in[12];
    const float* kw     = (const float*)d_in[13];
    const float* vw     = (const float*)d_in[14];
    const float* posw   = (const float*)d_in[15];
    const float* norm_g = (const float*)d_in[16];
    const float* norm_b = (const float*)d_in[17];
    const float* m1_w1  = (const float*)d_in[18];
    const float* m1_b1  = (const float*)d_in[19];
    const float* m1_g   = (const float*)d_in[20];
    const float* m1_be  = (const float*)d_in[21];
    const float* m1_w2  = (const float*)d_in[22];
    const float* m1_b2  = (const float*)d_in[23];
    float* out = (float*)d_out;

    char* ws = (char*)d_ws;
    const size_t MB = 1ull << 20;
    // small region [0,1MB)
    float* scores = (float*)(ws + 0);            // B*4096
    float* weff   = (float*)(ws + 160 * 1024);   // B*256
    int*   idx    = (int*)  (ws + 176 * 1024);   // B*1024
    float* p1sel  = (float*)(ws + 256 * 1024);   // B*3*1024
    // large regions with live-range aliasing
    float* Xsel = (float*)(ws + 1 * MB);    // 8MB  (B,1024,256)
    float* h    = (float*)(ws + 9 * MB);    // 8MB  (B,1024,256)
    float* Qb   = (float*)(ws + 1 * MB);    // 16MB (reuses Xsel+h)
    float* x2   = (float*)(ws + 17 * MB);   // 16MB (B,1024,512)
    float* Ob   = (float*)(ws + 17 * MB);   // 16MB (reuses x2)
    float* qin  = (float*)(ws + 33 * MB);   // 16MB
    float* o2   = (float*)(ws + 33 * MB);   // 16MB (reuses qin)
    float* kin  = (float*)(ws + 49 * MB);   // 16MB
    float* Kb   = (float*)(ws + 65 * MB);   // 16MB
    float* Vb   = (float*)(ws + 81 * MB);   // 16MB
    float* S    = (float*)(ws + 97 * MB);   // 32MB (B,1024,1024)

    const long long tokC  = (long long)N2 * Cin;   // 1024*256
    const long long tokO  = (long long)N2 * Cout;  // 1024*512
    const long long tokS  = (long long)N2 * N2;    // 1024*1024
    const float inv_scale = 1.0f / sqrtf((float)Cout);

    // --- selection path (folded q1/k1/e1/softmax/top_k) -------------------
    imff_stage1<<<B, 256, 0, stream>>>(l1, q1_w, k1_w, weff, Cin, Cout, N1);
    imff_stage2<<<(B * N1) / 256, 256, 0, stream>>>(l1, weff, scores, Cin, N1);
    imff_topk<<<B, 1024, 0, stream>>>(scores, idx, N1, N2);
    imff_gather_l1<<<(B * N2 * Cin) / 256, 256, 0, stream>>>(l1, idx, Xsel, Cin, N1, N2);
    imff_gather_xyz<<<(B * 3 * N2) / 256, 256, 0, stream>>>(xyz1, idx, p1sel, N1, N2);

    // --- point MLP: h = relu(LN(Xsel @ w1^T + b1)); x2 = h @ w2^T + b2 ----
    imff_gemm_wmma<false><<<dim3(Cin / TN, N2 / TM, B), 256, 0, stream>>>(
        Xsel, tokC, mlp_w1, 0, mlp_b1, h, tokC, N2, Cin, Cin, Cin, 1, 1.0f);
    imff_ln_act<<<B * N2, 256, 0, stream>>>(h, nullptr, mlp_g1, mlp_be1, Cin, N2, 1);
    imff_gemm_wmma<false><<<dim3(Cout / TN, N2 / TM, B), 256, 0, stream>>>(
        h, tokC, mlp_w2, 0, mlp_b2, x2, tokO, N2, Cout, Cin, Cout, 1, 1.0f);

    // --- positional inputs -------------------------------------------------
    imff_build_qin<<<B * N2, 256, 0, stream>>>(l2, xyz2, posw, qin, Cout, N2);
    imff_build_kin<<<B * N2, 256, 0, stream>>>(x2, p1sel, posw, kin, Cout, N2);

    // --- Q / K / V ---------------------------------------------------------
    imff_gemm_wmma<false><<<dim3(Cout / TN, N2 / TM, B), 256, 0, stream>>>(
        qin, tokO, qw, 0, nullptr, Qb, tokO, N2, Cout, Cout, Cout, 1, 1.0f);
    imff_gemm_wmma<false><<<dim3(Cout / TN, N2 / TM, B), 256, 0, stream>>>(
        kin, tokO, kw, 0, nullptr, Kb, tokO, N2, Cout, Cout, Cout, 1, 1.0f);
    imff_gemm_wmma<false><<<dim3(Cout / TN, N2 / TM, B), 256, 0, stream>>>(
        x2, tokO, vw, 0, nullptr, Vb, tokO, N2, Cout, Cout, Cout, 1, 1.0f);

    // --- attention: S = Q K^T / sqrt(Cout); softmax; O = A V --------------
    imff_gemm_wmma<false><<<dim3(N2 / TN, N2 / TM, B), 256, 0, stream>>>(
        Qb, tokO, Kb, tokO, nullptr, S, tokS, N2, N2, Cout, N2, 1, inv_scale);
    imff_softmax<<<B * N2, 256, 0, stream>>>(S, N2);
    imff_gemm_wmma<true><<<dim3(Cout / TN, N2 / TM, B), 256, 0, stream>>>(
        S, tokS, Vb, tokO, nullptr, Ob, tokO, N2, Cout, N2, Cout, 1, 1.0f);

    // --- o = leaky(LN(O + l2^T)); o = relu(LN(o @ m1_w1^T + b)); final ----
    imff_ln_act<<<B * N2, 256, 0, stream>>>(Ob, l2, norm_g, norm_b, Cout, N2, 2);
    imff_gemm_wmma<false><<<dim3(Cout / TN, N2 / TM, B), 256, 0, stream>>>(
        Ob, tokO, m1_w1, 0, m1_b1, o2, tokO, N2, Cout, Cout, Cout, 1, 1.0f);
    imff_ln_act<<<B * N2, 256, 0, stream>>>(o2, nullptr, m1_g, m1_be, Cout, N2, 1);
    // write transposed directly into d_out: out[b][r][tok]
    imff_gemm_wmma<false><<<dim3(Rout / TN, N2 / TM, B), 256, 0, stream>>>(
        o2, tokO, m1_w2, 0, m1_b2, out, (long long)Rout * N2,
        N2, Rout, Cout, 1, N2, 1.0f);
}
